// Attention_69801808495308
// MI455X (gfx1250) — matrix operand order repaired and checked by reading, not compiled, not run on hardware
//
#include <hip/hip_runtime.h>
#include <hip/hip_bf16.h>
#include <math.h>

typedef __attribute__((ext_vector_type(16))) _Float16 v16h;
typedef __attribute__((ext_vector_type(8)))  _Float16 v8h;
typedef __attribute__((ext_vector_type(8)))  float    v8f;

#define HEADS 8
#define QG    4
#define NTOK  2048
#define DIM   512
#define HD    64          // head dim
#define QN    (QG*NTOK)   // 8192 rows
#define LN_EPS 1e-5f

// ---------------------------------------------------------------------------
// helpers
// ---------------------------------------------------------------------------
__device__ __forceinline__ float warp_sum32(float v) {
#pragma unroll
  for (int o = 16; o > 0; o >>= 1) v += __shfl_xor(v, o, 32);
  return v;
}

// A-fragment (16x32 f16, row-major source, leading dim ld in halves).
// Per gfx1250 layout: lane L holds row M=L%16; halves j=0..15 map to
// K = j + 8*(j>=8) + 8*(L>=16)  -> two contiguous 8-half chunks at +0 / +16.
__device__ __forceinline__ v16h load_frag_a(const _Float16* __restrict__ base, int ld) {
  const int lane = threadIdx.x & 31;
  const _Float16* p = base + (size_t)(lane & 15) * ld + ((lane >> 4) << 3);
  v8h lo = *(const v8h*)(p);
  v8h hi = *(const v8h*)(p + 16);
  v16h r;
#pragma unroll
  for (int i = 0; i < 8; ++i) { r[i] = lo[i]; r[i + 8] = hi[i]; }
  return r;
}

// B-fragment (32x16 f16) loaded from a TRANSPOSED buffer Bt[col][k]
// (leading dim ld in halves). Lane L holds column N=L%16,
// K = j + 8*(j>=8) + 8*(L>=16) -> two contiguous 8-half chunks at +0 / +16.
__device__ __forceinline__ v16h load_frag_b(const _Float16* __restrict__ baseT, int ld) {
  const int lane = threadIdx.x & 31;
  const _Float16* p = baseT + (size_t)(lane & 15) * ld + ((lane >> 4) << 3);
  v8h lo = *(const v8h*)(p);
  v8h hi = *(const v8h*)(p + 16);
  v16h r;
#pragma unroll
  for (int i = 0; i < 8; ++i) { r[i] = lo[i]; r[i + 8] = hi[i]; }
  return r;
}

__device__ __forceinline__ v8f wmma_f16(v16h a, v16h b, v8f c) {
  return __builtin_amdgcn_wmma_f32_16x16x32_f16(false, a, false, b, (short)0, c,
                                                false, false);
}

// ---------------------------------------------------------------------------
// K0: fp32 -> f16 weight conversion (w layout [out][k] == Bt layout)
// ---------------------------------------------------------------------------
__global__ void k_cvt_f16(const float* __restrict__ w, _Float16* __restrict__ o, int n) {
  int i = blockIdx.x * blockDim.x + threadIdx.x;
  if (i < n) o[i] = (_Float16)w[i];
}

// ---------------------------------------------------------------------------
// K1: LayerNorm, one wave per row of 512
// ---------------------------------------------------------------------------
__global__ void k_ln(const float* __restrict__ x, const float* __restrict__ g,
                     const float* __restrict__ b, _Float16* __restrict__ o) {
  const int row  = blockIdx.x * 8 + (threadIdx.x >> 5);
  const int lane = threadIdx.x & 31;
  const float* xr = x + (size_t)row * DIM;
  float vals[16];
  float s = 0.f, s2 = 0.f;
#pragma unroll
  for (int i = 0; i < 16; ++i) {
    float v = xr[lane + 32 * i];
    vals[i] = v; s += v; s2 += v * v;
  }
  s  = warp_sum32(s);
  s2 = warp_sum32(s2);
  const float mu  = s * (1.f / DIM);
  const float var = s2 * (1.f / DIM) - mu * mu;
  const float inv = rsqrtf(var + LN_EPS);
  _Float16* orow = o + (size_t)row * DIM;
#pragma unroll
  for (int i = 0; i < 16; ++i) {
    int c = lane + 32 * i;
    orow[c] = (_Float16)((vals[i] - mu) * inv * g[c] + b[c]);
  }
}

// ---------------------------------------------------------------------------
// K2: generic f16 GEMM  C[M][N] = A[M][K] @ Bt[N][K]^T, f16 out
// block = 256 (8 waves), tile 64x64, wave -> (row tile, 2 col tiles)
// ---------------------------------------------------------------------------
__global__ void k_gemm_f16(const _Float16* __restrict__ A,
                           const _Float16* __restrict__ Bt,
                           _Float16* __restrict__ C, int N, int K) {
  const int bm   = blockIdx.x * 64;
  const int bn   = blockIdx.y * 64;
  const int w    = threadIdx.x >> 5;
  const int lane = threadIdx.x & 31;
  const int rt   = w & 3;
  const int ct0  = (w >> 2) * 2;

  const _Float16* Ab  = A  + (size_t)(bm + rt * 16) * K;
  const _Float16* Bb0 = Bt + (size_t)(bn + ct0 * 16) * K;
  const _Float16* Bb1 = Bt + (size_t)(bn + (ct0 + 1) * 16) * K;

  v8f acc0 = {}, acc1 = {};
  for (int k = 0; k < K; k += 32) {
    __builtin_prefetch(Ab + k + 256, 0, 1);
    v16h a  = load_frag_a(Ab  + k, K);
    v16h b0 = load_frag_b(Bb0 + k, K);
    v16h b1 = load_frag_b(Bb1 + k, K);
    acc0 = wmma_f16(a, b0, acc0);
    acc1 = wmma_f16(a, b1, acc1);
  }
  const int r0 = bm + rt * 16 + 8 * (lane >> 4);
  const int c0 = bn + ct0 * 16 + (lane & 15);
#pragma unroll
  for (int e = 0; e < 8; ++e) {
    C[(size_t)(r0 + e) * N + c0]      = (_Float16)acc0[e];
    C[(size_t)(r0 + e) * N + c0 + 16] = (_Float16)acc1[e];
  }
}

// ---------------------------------------------------------------------------
// K3: per-(head,row) stats on f16 features: 1/||x||, mean, unbiased var
// stats index layout: [h][row] with row = q*NTOK + n
// ---------------------------------------------------------------------------
__global__ void k_stats(const _Float16* __restrict__ F, float* __restrict__ invn,
                        float* __restrict__ mean, float* __restrict__ var) {
  int idx = blockIdx.x * blockDim.x + threadIdx.x;
  if (idx >= HEADS * QN) return;
  const int h = idx / QN;
  const int row = idx - h * QN;
  const _Float16* p = F + (size_t)row * DIM + h * HD;
  float s = 0.f, s2 = 0.f;
#pragma unroll
  for (int i = 0; i < HD; ++i) {
    float v = (float)p[i];
    s += v; s2 += v * v;
  }
  const float mu = s * (1.f / HD);
  invn[idx] = rsqrtf(fmaxf(s2, 1e-30f));
  mean[idx] = mu;
  var[idx]  = (s2 - (float)HD * mu * mu) * (1.f / (HD - 1));
}

// ---------------------------------------------------------------------------
// K4: build transposed operands for the M-GEMM.
// Kt2[h][q][128][NTOK]: rows 0..63 = invkn*K, rows 64..127 = K
// Vt [h][q][ 64][NTOK]
// ---------------------------------------------------------------------------
__global__ void k_build_t(const _Float16* __restrict__ FK,
                          const _Float16* __restrict__ FV,
                          const float* __restrict__ invkn,
                          _Float16* __restrict__ Kt2, _Float16* __restrict__ Vt) {
  long t = (long)blockIdx.x * blockDim.x + threadIdx.x;
  if (t >= (long)HEADS * QG * HD * NTOK) return;
  const int m = (int)(t % NTOK); long r = t / NTOK;
  const int i = (int)(r % HD);   r /= HD;
  const int q = (int)(r % QG);
  const int h = (int)(r / QG);
  const int row = q * NTOK + m;
  const float kv = (float)FK[(size_t)row * DIM + h * HD + i];
  const float vv = (float)FV[(size_t)row * DIM + h * HD + i];
  const float sc = invkn[h * QN + row];
  const size_t hq = (size_t)(h * QG + q);
  _Float16* kt = Kt2 + hq * 128 * NTOK;
  _Float16* vt = Vt  + hq * HD  * NTOK;
  kt[(size_t)i        * NTOK + m] = (_Float16)(kv * sc);
  kt[(size_t)(HD + i) * NTOK + m] = (_Float16)kv;
  vt[(size_t)i        * NTOK + m] = (_Float16)vv;
}

// ---------------------------------------------------------------------------
// K5: per (h,q) build [M1;M2] = Kt2(128xN) @ Vt(64xN)^T via WMMA; store
// TRANSPOSED (Mt[j][i], j=0..63, i=0..127) in f16 so it serves as B-frag
// source downstream. Also reduce u3 = sum kv[m] V[m][:], u4 = sum mk[m] V[m][:].
// block = 512 threads = 16 waves; 8 row-tiles x 4 col-tiles, 2 tiles/wave.
// ---------------------------------------------------------------------------
__global__ void k_build_M(const _Float16* __restrict__ Kt2,
                          const _Float16* __restrict__ Vt,
                          const float* __restrict__ kvar,
                          const float* __restrict__ kmean,
                          _Float16* __restrict__ Mtf, float* __restrict__ U) {
  const int hq = blockIdx.x;
  const _Float16* A  = Kt2 + (size_t)hq * 128 * NTOK;
  const _Float16* Bt = Vt  + (size_t)hq * HD  * NTOK;
  const int w    = threadIdx.x >> 5;
  const int lane = threadIdx.x & 31;
  const int rt   = w & 7;
  const int ct0  = (w >> 3) * 2;

  const _Float16* Ab  = A  + (size_t)rt * 16 * NTOK;
  const _Float16* Bb0 = Bt + (size_t)ct0 * 16 * NTOK;
  const _Float16* Bb1 = Bt + (size_t)(ct0 + 1) * 16 * NTOK;

  v8f acc0 = {}, acc1 = {};
  for (int k = 0; k < NTOK; k += 32) {
    __builtin_prefetch(Ab + k + 256, 0, 1);
    v16h a  = load_frag_a(Ab  + k, NTOK);
    v16h b0 = load_frag_b(Bb0 + k, NTOK);
    v16h b1 = load_frag_b(Bb1 + k, NTOK);
    acc0 = wmma_f16(a, b0, acc0);
    acc1 = wmma_f16(a, b1, acc1);
  }
  _Float16* Mo = Mtf + (size_t)hq * 64 * 128;
  const int i0 = rt * 16 + 8 * (lane >> 4);
  const int j0 = ct0 * 16 + (lane & 15);
#pragma unroll
  for (int e = 0; e < 8; ++e) {
    Mo[(size_t)j0        * 128 + i0 + e] = (_Float16)acc0[e];
    Mo[(size_t)(j0 + 16) * 128 + i0 + e] = (_Float16)acc1[e];
  }

  // u3[j] (tid 0..63) and u4[j] (tid 64..127)
  if (threadIdx.x < 128) {
    const int h = hq / QG, q = hq - h * QG;
    const int j = threadIdx.x & 63;
    const float* st = ((threadIdx.x < 64) ? kvar : kmean) + h * QN + q * NTOK;
    const _Float16* vrow = Bt + (size_t)j * NTOK;
    float s = 0.f;
    for (int m = 0; m < NTOK; ++m) s += st[m] * (float)vrow[m];
    U[(size_t)hq * 128 + threadIdx.x] = s;
  }
}

// ---------------------------------------------------------------------------
// K6: apply — out_mid[n][64h+j] = cos_w*invqn[n]*(q@M1) + (cov_w/d)*(q@M2)
//                                + (var_w/d)*qvar[n]*u3[j] - cov_w*qmean[n]*u4[j]
// grid (32 hq, 16 n-chunks of 128), block 256 = 8 waves; wave = 16-row tile,
// all 4 col tiles, dual accumulators (M1/M2).
// ---------------------------------------------------------------------------
__global__ void k_apply(const _Float16* __restrict__ FQ,
                        const _Float16* __restrict__ Mtf,
                        const float* __restrict__ U,
                        const float* __restrict__ invqn,
                        const float* __restrict__ qmean,
                        const float* __restrict__ qvar,
                        const float* __restrict__ cov_raw,
                        const float* __restrict__ var_raw,
                        _Float16* __restrict__ Omid) {
  const int hq = blockIdx.x;
  const int h = hq / QG, q = hq - h * QG;
  const int n0 = blockIdx.y * 128;
  const int w = threadIdx.x >> 5;
  const int lane = threadIdx.x & 31;

  const float cov_w = 1.f / (1.f + __expf(-cov_raw[0]));
  const float var_w = 1.f / (1.f + __expf(-var_raw[0]));
  const float cos_w = 1.f - cov_w - var_w;

  const _Float16* Aq = FQ + (size_t)(q * NTOK + n0 + w * 16) * DIM + h * HD;
  const _Float16* Mo = Mtf + (size_t)hq * 64 * 128;

  v8f a1[4] = {}, a2[4] = {};
  for (int kk = 0; kk < HD; kk += 32) {
    v16h a = load_frag_a(Aq + kk, DIM);
#pragma unroll
    for (int t = 0; t < 4; ++t) {
      v16h b1 = load_frag_b(Mo + (size_t)t * 16 * 128 + kk, 128);       // M1 part
      v16h b2 = load_frag_b(Mo + (size_t)t * 16 * 128 + HD + kk, 128);  // M2 part
      a1[t] = wmma_f16(a, b1, a1[t]);
      a2[t] = wmma_f16(a, b2, a2[t]);
    }
  }

  const float* Uv = U + (size_t)hq * 128;
  const int rbase = n0 + w * 16 + 8 * (lane >> 4);
#pragma unroll
  for (int e = 0; e < 8; ++e) {
    const int n = rbase + e;
    const int si = h * QN + q * NTOK + n;
    const float iqn = invqn[si], mq = qmean[si], qv = qvar[si];
#pragma unroll
    for (int t = 0; t < 4; ++t) {
      const int j = t * 16 + (lane & 15);
      float val = cos_w * iqn * a1[t][e] + (cov_w * (1.f / HD)) * a2[t][e]
                + (var_w * (1.f / HD)) * qv * Uv[j] - cov_w * mq * Uv[64 + j];
      Omid[(size_t)(q * NTOK + n) * DIM + h * HD + j] = (_Float16)val;
    }
  }
}

// ---------------------------------------------------------------------------
// K7: final GEMM with bias, fp32 output: out = Omid @ w_out^T + b_out
// ---------------------------------------------------------------------------
__global__ void k_gemm_out(const _Float16* __restrict__ A,
                           const _Float16* __restrict__ Bt,
                           const float* __restrict__ bias,
                           float* __restrict__ C) {
  const int bm   = blockIdx.x * 64;
  const int bn   = blockIdx.y * 64;
  const int w    = threadIdx.x >> 5;
  const int lane = threadIdx.x & 31;
  const int rt   = w & 3;
  const int ct0  = (w >> 2) * 2;

  const _Float16* Ab  = A  + (size_t)(bm + rt * 16) * DIM;
  const _Float16* Bb0 = Bt + (size_t)(bn + ct0 * 16) * DIM;
  const _Float16* Bb1 = Bt + (size_t)(bn + (ct0 + 1) * 16) * DIM;

  v8f acc0 = {}, acc1 = {};
  for (int k = 0; k < DIM; k += 32) {
    __builtin_prefetch(Ab + k + 256, 0, 1);
    v16h a  = load_frag_a(Ab  + k, DIM);
    v16h b0 = load_frag_b(Bb0 + k, DIM);
    v16h b1 = load_frag_b(Bb1 + k, DIM);
    acc0 = wmma_f16(a, b0, acc0);
    acc1 = wmma_f16(a, b1, acc1);
  }
  const int r0 = bm + rt * 16 + 8 * (lane >> 4);
  const int c0 = bn + ct0 * 16 + (lane & 15);
#pragma unroll
  for (int e = 0; e < 8; ++e) {
    C[(size_t)(r0 + e) * DIM + c0]      = acc0[e] + bias[c0];
    C[(size_t)(r0 + e) * DIM + c0 + 16] = acc1[e] + bias[c0 + 16];
  }
}

// ---------------------------------------------------------------------------
// launch
// ---------------------------------------------------------------------------
extern "C" void kernel_launch(void* const* d_in, const int* in_sizes, int n_in,
                              void* d_out, int out_size, void* d_ws, size_t ws_size,
                              hipStream_t stream) {
  const float* q   = (const float*)d_in[0];
  const float* k   = (const float*)d_in[1];
  const float* v   = (const float*)d_in[2];
  const float* lg  = (const float*)d_in[3];
  const float* lb  = (const float*)d_in[4];
  const float* win = (const float*)d_in[5];
  const float* wout= (const float*)d_in[6];
  const float* bout= (const float*)d_in[7];
  const float* cwr = (const float*)d_in[8];
  const float* vwr = (const float*)d_in[9];
  float* out = (float*)d_out;

  char* ws = (char*)d_ws;
  size_t off = 0;
  auto carve = [&](size_t bytes) {
    void* p = ws + off;
    off += (bytes + 255) & ~(size_t)255;
    return p;
  };
  _Float16* WinH  = (_Float16*)carve((size_t)DIM * DIM * 2);
  _Float16* WoutH = (_Float16*)carve((size_t)DIM * DIM * 2);
  _Float16* Xln   = (_Float16*)carve((size_t)QN * DIM * 2);
  _Float16* FQ    = (_Float16*)carve((size_t)QN * DIM * 2);
  _Float16* FK    = (_Float16*)carve((size_t)QN * DIM * 2);
  _Float16* FV    = (_Float16*)carve((size_t)QN * DIM * 2);
  float* invqn = (float*)carve((size_t)HEADS * QN * 4);
  float* qmean = (float*)carve((size_t)HEADS * QN * 4);
  float* qvar  = (float*)carve((size_t)HEADS * QN * 4);
  float* invkn = (float*)carve((size_t)HEADS * QN * 4);
  float* kmean = (float*)carve((size_t)HEADS * QN * 4);
  float* kvar  = (float*)carve((size_t)HEADS * QN * 4);
  _Float16* Kt2 = (_Float16*)carve((size_t)HEADS * QG * 128 * NTOK * 2);
  _Float16* Vt  = (_Float16*)carve((size_t)HEADS * QG * HD  * NTOK * 2);
  _Float16* Mtf = (_Float16*)carve((size_t)HEADS * QG * 64 * 128 * 2);
  float*    U   = (float*)carve((size_t)HEADS * QG * 128 * 4);
  _Float16* Omid= (_Float16*)carve((size_t)QN * DIM * 2);
  (void)ws_size; (void)n_in; (void)in_sizes; (void)out_size;

  const int wn = DIM * DIM;
  k_cvt_f16<<<(wn + 255) / 256, 256, 0, stream>>>(win,  WinH,  wn);
  k_cvt_f16<<<(wn + 255) / 256, 256, 0, stream>>>(wout, WoutH, wn);

  dim3 gGemm(QN / 64, DIM / 64);
  // q / k / v projections (LN buffer reused sequentially)
  const float* src[3] = {q, k, v};
  _Float16* dst[3] = {FQ, FK, FV};
  for (int t = 0; t < 3; ++t) {
    k_ln<<<QN / 8, 256, 0, stream>>>(src[t], lg, lb, Xln);
    k_gemm_f16<<<gGemm, 256, 0, stream>>>(Xln, WinH, dst[t], DIM, DIM);
  }

  k_stats<<<(HEADS * QN) / 256, 256, 0, stream>>>(FQ, invqn, qmean, qvar);
  k_stats<<<(HEADS * QN) / 256, 256, 0, stream>>>(FK, invkn, kmean, kvar);

  const long ttot = (long)HEADS * QG * HD * NTOK;
  k_build_t<<<(unsigned)((ttot + 255) / 256), 256, 0, stream>>>(FK, FV, invkn, Kt2, Vt);

  k_build_M<<<HEADS * QG, 512, 0, stream>>>(Kt2, Vt, kvar, kmean, Mtf, U);

  dim3 gApply(HEADS * QG, NTOK / 128);
  k_apply<<<gApply, 256, 0, stream>>>(FQ, Mtf, U, invqn, qmean, qvar, cwr, vwr, Omid);

  k_gemm_out<<<gGemm, 256, 0, stream>>>(Omid, WoutH, bout, out);
}
